// SlowAR_64476049047591
// MI455X (gfx1250) — matrix operand order repaired and checked by reading, not compiled, not run on hardware
//
#include <hip/hip_runtime.h>
#include <hip/hip_bf16.h>

// ---------------- model constants ----------------
enum : int { Tt = 2048, Dd = 1024, Hh = 16, HKVc = 4, HDc = 64,
             Ec = 8, Fc = 2048, Vc = 32000, Lc = 2,
             WINDOW_C = 1024, NGLOBAL_C = 32 };

typedef __attribute__((ext_vector_type(16))) __bf16        v16bf;
typedef __attribute__((ext_vector_type(2)))  __bf16        v2bf;
typedef __attribute__((ext_vector_type(8)))  float         v8f;
typedef __attribute__((ext_vector_type(4)))  unsigned int  v4u;

struct V16Pair { v4u lo; v4u hi; };   // 32 bytes, no padding

// float->bf16 RNE via compiler lowering (uses native cvt if gfx1250 has one)
__device__ __forceinline__ unsigned int pack_bf16x2(float a, float b) {
  v2bf r = { (__bf16)a, (__bf16)b };
  return __builtin_bit_cast(unsigned int, r);
}
__device__ __forceinline__ unsigned short f32_to_bf16u(float a) {
  return __builtin_bit_cast(unsigned short, (__bf16)a);
}
__device__ __forceinline__ float bf16u_to_f(unsigned short u) {
  return __builtin_bit_cast(float, ((unsigned int)u) << 16);
}

// ---------------- WMMA GEMM (bf16 in, fp32 accumulate) ----------------
// C[M,N] = alpha * A[M,K] @ B   (A,B bf16; B: KxN row-major, or NxK row-major if BT)
// OM=0: C fp32 store.  OM=1: C fp32 += rowscale[i*rs_stride]*alpha*(..) (null->1).
// OM=2: C bf16 store.
// Block tile 256x64, 8 waves in 4x2 grid, each wave 64x32 (4x2 fragments).
// Double-buffered LDS: next tile's global loads overlap current tile's 8 WMMAs.
// REQUIRES: M%256==0, N%64==0, K%32==0, lda/ldb%8==0 (true for every GEMM here)
#define BM 256
#define BN 64
#define BKK 32
#define LDS_STRIDE 40   // ushorts per row: 32 data + 8 pad (80B rows, 16B aligned)

template<bool BT, int OM>
__global__ __launch_bounds__(256)
void gemm_wmma(const unsigned short* __restrict__ A, int lda,
               const unsigned short* __restrict__ B, int ldb,
               void* __restrict__ Cv, int ldc,
               int M, int N, int K, float alpha,
               const float* __restrict__ rowscale, int rs_stride) {
  // sA rows natural K order (A fragment split {0-15|16-31}), lane's A fragment contiguous (32B).
  // sB transposed [n][k], natural K order (B fragment split {0-15|16-31}).
  __shared__ __align__(16) unsigned short sA[2][BM * LDS_STRIDE];
  __shared__ __align__(16) unsigned short sB[2][BN * LDS_STRIDE];

  const int tid  = threadIdx.x;
  const int lane = tid & 31;
  const int wave = tid >> 5;
  const int wm = wave & 3;    // 64-row block
  const int wn = wave >> 2;   // 32-col block
  const int row0 = blockIdx.y * BM;
  const int col0 = blockIdx.x * BN;
  const int mfrag = lane & 15;
  const int half  = lane >> 4;

  // staging coordinates (A: 4 chunks of 8 bf16; B: 1 chunk)
  const int ar  = tid >> 2;            // base row for A chunk p: ar + p*64
  const int ac8 = (tid & 3) * 8;       // K group base
  const int aj0 = ac8;  // natural K
  const int btn  = tid >> 2;           // BT: n
  const int btc8 = (tid & 3) * 8;      // BT: k group
  const int bnk  = tid >> 3;           // normal: k
  const int bnn8 = (tid & 7) * 8;      // normal: n group

  v8f acc[4][2] = {};

  auto loadA = [&](int k0, v4u va[4]) {
#pragma unroll
    for (int p = 0; p < 4; ++p)
      va[p] = *(const v4u*)(A + (size_t)(row0 + ar + p * 64) * lda + k0 + ac8);
  };
  auto storeA = [&](int buf, const v4u va[4]) {
#pragma unroll
    for (int p = 0; p < 4; ++p)
      *(v4u*)&sA[buf][(ar + p * 64) * LDS_STRIDE + aj0] = va[p];
  };
  auto loadB = [&](int k0, v4u& vb) {
    if (BT) vb = *(const v4u*)(B + (size_t)(col0 + btn) * ldb + k0 + btc8);
    else    vb = *(const v4u*)(B + (size_t)(k0 + bnk) * ldb + col0 + bnn8);
  };
  auto storeB = [&](int buf, const v4u& vb) {
    if (BT) {
      *(v4u*)&sB[buf][btn * LDS_STRIDE + btc8] = vb;
    } else {
#pragma unroll
      for (int j = 0; j < 4; ++j) {
        unsigned int w = vb[j];
        sB[buf][(bnn8 + 2 * j + 0) * LDS_STRIDE + bnk] = (unsigned short)(w & 0xffffu);
        sB[buf][(bnn8 + 2 * j + 1) * LDS_STRIDE + bnk] = (unsigned short)(w >> 16);
      }
    }
  };

  // ---- prologue: stage tile 0 ----
  {
    v4u va[4]; v4u vb;
    loadA(0, va); loadB(0, vb);
    storeA(0, va); storeB(0, vb);
  }
  __syncthreads();

  const int nk = K / BKK;
  for (int it = 0; it < nk; ++it) {
    const int cur = it & 1, nxt = cur ^ 1;
    const bool more = (it + 1 < nk);
    v4u va[4]; v4u vb;
    if (more) { loadA((it + 1) * BKK, va); loadB((it + 1) * BKK, vb); }
    if (it + 2 < nk)
      __builtin_prefetch(A + (size_t)(row0 + (tid >> 1)) * lda + (it + 2) * BKK, 0, 1);

    // ---- compute from current buffer ----
    v16bf afr[4], bfr[2];
#pragma unroll
    for (int ti = 0; ti < 4; ++ti) {
      const unsigned short* pa =
          &sA[cur][(wm * 64 + ti * 16 + mfrag) * LDS_STRIDE + half * 16];
      V16Pair t; t.lo = *(const v4u*)pa; t.hi = *(const v4u*)(pa + 8);
      afr[ti] = __builtin_bit_cast(v16bf, t);
    }
#pragma unroll
    for (int tj = 0; tj < 2; ++tj) {
      const unsigned short* pb =
          &sB[cur][(wn * 32 + tj * 16 + mfrag) * LDS_STRIDE + half * 16];
      V16Pair t; t.lo = *(const v4u*)pb; t.hi = *(const v4u*)(pb + 8);
      bfr[tj] = __builtin_bit_cast(v16bf, t);
    }
#pragma unroll
    for (int ti = 0; ti < 4; ++ti)
#pragma unroll
      for (int tj = 0; tj < 2; ++tj)
        acc[ti][tj] = __builtin_amdgcn_wmma_f32_16x16x32_bf16(
            false, afr[ti], false, bfr[tj], (short)0, acc[ti][tj], false, false);

    // ---- stage next tile into alternate buffer ----
    if (more) { storeA(nxt, va); storeB(nxt, vb); }
    __syncthreads();
  }

  // ---- epilogue ----
  float* Cf = (float*)Cv;
  unsigned short* Cb = (unsigned short*)Cv;
#pragma unroll
  for (int ti = 0; ti < 4; ++ti) {
#pragma unroll
    for (int tj = 0; tj < 2; ++tj) {
      int colg = col0 + wn * 32 + tj * 16 + mfrag;
#pragma unroll
      for (int r = 0; r < 8; ++r) {
        int rowg = row0 + wm * 64 + ti * 16 + r + (half << 3);
        float v = acc[ti][tj][r] * alpha;
        if (OM == 1) {
          float s = rowscale ? rowscale[(size_t)rowg * rs_stride] : 1.0f;
          Cf[(size_t)rowg * ldc + colg] += s * v;
        } else if (OM == 2) {
          Cb[(size_t)rowg * ldc + colg] = f32_to_bf16u(v);
        } else {
          Cf[(size_t)rowg * ldc + colg] = v;
        }
      }
    }
  }
}

// ---------------- conversion / elementwise / reduction kernels ----------------
__global__ void cvt_bf16_kernel(const float* __restrict__ S,
                                unsigned short* __restrict__ Dst, int n4) {
  int i = blockIdx.x * blockDim.x + threadIdx.x;
  if (i >= n4) return;
  const float4 v = ((const float4*)S)[i];
  uint2 u; u.x = pack_bf16x2(v.x, v.y); u.y = pack_bf16x2(v.z, v.w);
  ((uint2*)Dst)[i] = u;
}

__global__ void embed_kernel(const int* __restrict__ tok,
                             const float* __restrict__ emb,
                             float* __restrict__ X) {
  int t = blockIdx.x;
  int id = tok[t];
  for (int d = threadIdx.x; d < Dd; d += blockDim.x)
    X[(size_t)t * Dd + d] = emb[(size_t)id * Dd + d];
}

__global__ __launch_bounds__(256)
void rmsnorm_kernel(const float* __restrict__ X, const float* __restrict__ W,
                    unsigned short* __restrict__ Y) {
  int row = blockIdx.x, tid = threadIdx.x;
  const float4 v = *(const float4*)(X + (size_t)row * Dd + tid * 4);
  float ss = v.x * v.x + v.y * v.y + v.z * v.z + v.w * v.w;
  __shared__ float red[256];
  red[tid] = ss; __syncthreads();
  for (int s = 128; s > 0; s >>= 1) { if (tid < s) red[tid] += red[tid + s]; __syncthreads(); }
  float inv = rsqrtf(red[0] / (float)Dd + 1.1920929e-7f);
  const float4 w = *(const float4*)(W + tid * 4);
  uint2 u;
  u.x = pack_bf16x2(v.x * inv * w.x, v.y * inv * w.y);
  u.y = pack_bf16x2(v.z * inv * w.z, v.w * inv * w.w);
  *(uint2*)(Y + (size_t)row * Dd + tid * 4) = u;
}

__global__ void rope_kernel(const float* __restrict__ X, unsigned short* __restrict__ Xb,
                            int nheads) {
  int idx = blockIdx.x * blockDim.x + threadIdx.x;
  int total = Tt * nheads * 32;
  if (idx >= total) return;
  int d = idx & 31;
  int hh = (idx >> 5) % nheads;
  int t = idx / (nheads * 32);
  float invf = __expf(-((float)(2 * d) / 64.0f) * __logf(500000.0f));
  float ang = (float)t * invf;
  float c = __cosf(ang), s = __sinf(ang);
  size_t base = (size_t)t * (nheads * 64) + hh * 64 + d;
  float x1 = X[base], x2 = X[base + 32];
  Xb[base]      = f32_to_bf16u(x1 * c - x2 * s);
  Xb[base + 32] = f32_to_bf16u(x2 * c + x1 * s);
}

__global__ __launch_bounds__(256)
void softmax_mask_kernel(const float* __restrict__ S, unsigned short* __restrict__ P) {
  int i = blockIdx.x, tid = threadIdx.x;
  const float* row = S + (size_t)i * Tt;
  float v[Tt / 256];
  float mx = -3.4e38f;
#pragma unroll
  for (int p = 0; p < Tt / 256; ++p) {
    int j = tid + p * 256;
    bool allowed = (j <= i) && ((i - j < WINDOW_C) || (j < NGLOBAL_C));
    float s = allowed ? row[j] : -3.4e38f;
    v[p] = s; mx = fmaxf(mx, s);
  }
  __shared__ float red[256];
  red[tid] = mx; __syncthreads();
  for (int s = 128; s > 0; s >>= 1) { if (tid < s) red[tid] = fmaxf(red[tid], red[tid + s]); __syncthreads(); }
  mx = red[0]; __syncthreads();
  float sum = 0.f;
#pragma unroll
  for (int p = 0; p < Tt / 256; ++p) {
    float e = (v[p] > -1.0e37f) ? __expf(v[p] - mx) : 0.0f;
    v[p] = e; sum += e;
  }
  red[tid] = sum; __syncthreads();
  for (int s = 128; s > 0; s >>= 1) { if (tid < s) red[tid] += red[tid + s]; __syncthreads(); }
  float inv = 1.0f / red[0];
#pragma unroll
  for (int p = 0; p < Tt / 256; ++p)
    P[(size_t)i * Tt + tid + p * 256] = f32_to_bf16u(v[p] * inv);
}

__global__ void router_kernel(const unsigned short* __restrict__ H2,
                              const float* __restrict__ RW,
                              float* __restrict__ probs, float* __restrict__ wts) {
  int t = blockIdx.x * blockDim.x + threadIdx.x;
  if (t >= Tt) return;
  float acc[Ec] = {};
  const unsigned short* h = H2 + (size_t)t * Dd;
  for (int d = 0; d < Dd; ++d) {
    float hv = bf16u_to_f(h[d]);
#pragma unroll
    for (int e = 0; e < Ec; ++e) acc[e] += hv * RW[d * Ec + e];
  }
  float mx = acc[0];
#pragma unroll
  for (int e = 1; e < Ec; ++e) mx = fmaxf(mx, acc[e]);
  float sum = 0.f;
#pragma unroll
  for (int e = 0; e < Ec; ++e) { acc[e] = __expf(acc[e] - mx); sum += acc[e]; }
  float invs = 1.0f / sum;
#pragma unroll
  for (int e = 0; e < Ec; ++e) { acc[e] *= invs; probs[t * Ec + e] = acc[e]; }
  int i1 = 0;
#pragma unroll
  for (int e = 1; e < Ec; ++e) if (acc[e] > acc[i1]) i1 = e;
  int i2 = (i1 == 0) ? 1 : 0;
#pragma unroll
  for (int e = 0; e < Ec; ++e) if (e != i1 && acc[e] > acc[i2]) i2 = e;
  float wsum = acc[i1] + acc[i2];
#pragma unroll
  for (int e = 0; e < Ec; ++e)
    wts[t * Ec + e] = (e == i1) ? acc[i1] / wsum : ((e == i2) ? acc[i2] / wsum : 0.0f);
}

__global__ __launch_bounds__(256)
void aux_kernel(const float* __restrict__ probs, const float* __restrict__ wts,
                float* __restrict__ auxacc) {
  __shared__ float sp[256 * Ec];
  __shared__ float sf[256 * Ec];
  int tid = threadIdx.x;
  float lp[Ec] = {}, lf[Ec] = {};
  for (int t = tid; t < Tt; t += 256)
#pragma unroll
    for (int e = 0; e < Ec; ++e) {
      lp[e] += probs[t * Ec + e];
      lf[e] += (wts[t * Ec + e] > 0.0f) ? 1.0f : 0.0f;
    }
#pragma unroll
  for (int e = 0; e < Ec; ++e) { sp[tid * Ec + e] = lp[e]; sf[tid * Ec + e] = lf[e]; }
  __syncthreads();
  for (int s = 128; s > 0; s >>= 1) {
    if (tid < s)
#pragma unroll
      for (int e = 0; e < Ec; ++e) {
        sp[tid * Ec + e] += sp[(tid + s) * Ec + e];
        sf[tid * Ec + e] += sf[(tid + s) * Ec + e];
      }
    __syncthreads();
  }
  if (tid == 0) {
    float a = 0.f;
#pragma unroll
    for (int e = 0; e < Ec; ++e) a += (sp[e] / (float)Tt) * (sf[e] / (float)Tt);
    auxacc[0] += (float)Ec * a;
  }
}

__global__ void swiglu_kernel(const float* __restrict__ G, const float* __restrict__ U,
                              unsigned short* __restrict__ Ab, int n) {
  int i = blockIdx.x * blockDim.x + threadIdx.x;
  if (i < n) {
    float g = G[i];
    Ab[i] = f32_to_bf16u((g / (1.0f + __expf(-g))) * U[i]);
  }
}

__global__ void zero_kernel(float* p, int n) {
  int i = blockIdx.x * blockDim.x + threadIdx.x;
  if (i < n) p[i] = 0.0f;
}

__global__ void write_aux_kernel(const float* __restrict__ auxacc, float* __restrict__ out) {
  out[0] = auxacc[0];
}

// ---------------- host orchestration ----------------
extern "C" void kernel_launch(void* const* d_in, const int* in_sizes, int n_in,
                              void* d_out, int out_size, void* d_ws, size_t ws_size,
                              hipStream_t stream) {
  (void)in_sizes; (void)n_in; (void)out_size; (void)ws_size;
  const int*   tokens = (const int*)d_in[0];
  const float* embedw = (const float*)d_in[1];
  const float* norm1  = (const float*)d_in[2];
  const float* norm2  = (const float*)d_in[3];
  const float* wq     = (const float*)d_in[4];
  const float* wk     = (const float*)d_in[5];
  const float* wv     = (const float*)d_in[6];
  const float* wo     = (const float*)d_in[7];
  const float* rw     = (const float*)d_in[8];
  const float* wg     = (const float*)d_in[9];
  const float* wu     = (const float*)d_in[10];
  const float* wd     = (const float*)d_in[11];
  const float* normo  = (const float*)d_in[12];
  float* out = (float*)d_out;

  // ---- workspace carve-out ----
  char* ws = (char*)d_ws;
  size_t off = 0;
  auto allocB = [&](size_t nbytes) -> void* {
    void* p = (void*)(ws + off);
    off += nbytes;
    off = (off + 255) & ~(size_t)255;
    return p;
  };
  float* x      = (float*)allocB((size_t)Tt * Dd * 4);            // residual (fp32)
  float* kfp    = (float*)allocB((size_t)Tt * HKVc * HDc * 4);    // K pre-rope (fp32)
  float* gate   = (float*)allocB((size_t)Tt * Fc * 4);            // fp32; aliases scores
  float* upb    = (float*)allocB((size_t)Tt * Fc * 4);            // fp32; aliases q (pre-rope)
  unsigned short* hb   = (unsigned short*)allocB((size_t)Tt * Dd * 2);        // normed (bf16)
  unsigned short* qb   = (unsigned short*)allocB((size_t)Tt * Hh * HDc * 2);  // roped Q (bf16)
  unsigned short* kb   = (unsigned short*)allocB((size_t)Tt * HKVc * HDc * 2);
  unsigned short* vb   = (unsigned short*)allocB((size_t)Tt * HKVc * HDc * 2);
  unsigned short* ab   = (unsigned short*)allocB((size_t)Tt * Hh * HDc * 2);  // attn out (bf16)
  unsigned short* actb = (unsigned short*)allocB((size_t)Tt * Fc * 2);        // aliases pb
  unsigned short* wbuf = (unsigned short*)allocB((size_t)Vc * Dd * 2);        // bf16 weight scratch
  float* probs  = (float*)allocB((size_t)Tt * Ec * 4);
  float* wts    = (float*)allocB((size_t)Tt * Ec * 4);
  float* auxacc = (float*)allocB(4);
  float* scores = gate;                // time-disjoint alias (Tt*Tt == Tt*Fc)
  float* qfp    = upb;                 // Q pre-rope (8MB) lives in upb (16MB)
  unsigned short* pb = actb;           // bf16 attn probabilities

  dim3 blk(256);
  auto cvt = [&](const float* src, unsigned short* dst, size_t n) {
    int n4 = (int)(n / 4);
    cvt_bf16_kernel<<<(n4 + 255) / 256, 256, 0, stream>>>(src, dst, n4);
  };
  auto gemm = [&](const unsigned short* A, int lda, const unsigned short* B, int ldb,
                  float* C, int ldc, int M, int N, int K, float alpha) {
    dim3 g(N / BN, M / BM);
    gemm_wmma<false, 0><<<g, blk, 0, stream>>>(A, lda, B, ldb, (void*)C, ldc, M, N, K, alpha,
                                               (const float*)nullptr, 0);
  };
  auto gemmBT = [&](const unsigned short* A, int lda, const unsigned short* B, int ldb,
                    float* C, int ldc, int M, int N, int K, float alpha) {
    dim3 g(N / BN, M / BM);
    gemm_wmma<true, 0><<<g, blk, 0, stream>>>(A, lda, B, ldb, (void*)C, ldc, M, N, K, alpha,
                                              (const float*)nullptr, 0);
  };
  auto gemmACC = [&](const unsigned short* A, int lda, const unsigned short* B, int ldb,
                     float* C, int ldc, int M, int N, int K, float alpha,
                     const float* rs, int rs_stride) {
    dim3 g(N / BN, M / BM);
    gemm_wmma<false, 1><<<g, blk, 0, stream>>>(A, lda, B, ldb, (void*)C, ldc, M, N, K, alpha,
                                               rs, rs_stride);
  };
  auto gemmBF = [&](const unsigned short* A, int lda, const unsigned short* B, int ldb,
                    unsigned short* C, int ldc, int M, int N, int K, float alpha) {
    dim3 g(N / BN, M / BM);
    gemm_wmma<false, 2><<<g, blk, 0, stream>>>(A, lda, B, ldb, (void*)C, ldc, M, N, K, alpha,
                                               (const float*)nullptr, 0);
  };

  zero_kernel<<<1, 32, 0, stream>>>(auxacc, 1);
  embed_kernel<<<Tt, 256, 0, stream>>>(tokens, embedw, x);

  for (int l = 0; l < Lc; ++l) {
    // --- attention ---
    rmsnorm_kernel<<<Tt, 256, 0, stream>>>(x, norm1 + (size_t)l * Dd, hb);
    cvt(wq + (size_t)l * Dd * (Hh * HDc), wbuf, (size_t)Dd * Hh * HDc);
    gemm(hb, Dd, wbuf, Hh * HDc, qfp, Hh * HDc, Tt, Hh * HDc, Dd, 1.0f);
    cvt(wk + (size_t)l * Dd * (HKVc * HDc), wbuf, (size_t)Dd * HKVc * HDc);
    gemm(hb, Dd, wbuf, HKVc * HDc, kfp, HKVc * HDc, Tt, HKVc * HDc, Dd, 1.0f);
    cvt(wv + (size_t)l * Dd * (HKVc * HDc), wbuf, (size_t)Dd * HKVc * HDc);
    gemmBF(hb, Dd, wbuf, HKVc * HDc, vb, HKVc * HDc, Tt, HKVc * HDc, Dd, 1.0f);
    rope_kernel<<<(Tt * Hh * 32 + 255) / 256, 256, 0, stream>>>(qfp, qb, Hh);
    rope_kernel<<<(Tt * HKVc * 32 + 255) / 256, 256, 0, stream>>>(kfp, kb, HKVc);
    for (int hh = 0; hh < Hh; ++hh) {
      int kv = hh >> 2;  // GQA group
      gemm_wmma<true, 0><<<dim3(Tt / BN, Tt / BM), blk, 0, stream>>>(
          qb + hh * HDc, Hh * HDc, kb + kv * HDc, HKVc * HDc,
          (void*)scores, Tt, Tt, Tt, HDc, 0.125f, (const float*)nullptr, 0);  // QK^T * HD^-0.5
      softmax_mask_kernel<<<Tt, 256, 0, stream>>>(scores, pb);
      gemmBF(pb, Tt, vb + kv * HDc, HKVc * HDc,
             ab + hh * HDc, Hh * HDc, Tt, HDc, Tt, 1.0f);                     // P @ V -> bf16
    }
    cvt(wo + (size_t)l * (Hh * HDc) * Dd, wbuf, (size_t)Hh * HDc * Dd);
    gemmACC(ab, Hh * HDc, wbuf, Dd, x, Dd, Tt, Dd, Hh * HDc, 1.0f, nullptr, 0);  // x += attn@Wo

    // --- MoE ---
    rmsnorm_kernel<<<Tt, 256, 0, stream>>>(x, norm2 + (size_t)l * Dd, hb);
    router_kernel<<<(Tt + 255) / 256, 256, 0, stream>>>(hb, rw + (size_t)l * Dd * Ec, probs, wts);
    aux_kernel<<<1, 256, 0, stream>>>(probs, wts, auxacc);
    for (int e = 0; e < Ec; ++e) {
      cvt(wg + ((size_t)l * Ec + e) * Dd * Fc, wbuf, (size_t)Dd * Fc);
      gemm(hb, Dd, wbuf, Fc, gate, Fc, Tt, Fc, Dd, 1.0f);
      cvt(wu + ((size_t)l * Ec + e) * Dd * Fc, wbuf, (size_t)Dd * Fc);
      gemm(hb, Dd, wbuf, Fc, upb, Fc, Tt, Fc, Dd, 1.0f);
      swiglu_kernel<<<(Tt * Fc + 255) / 256, 256, 0, stream>>>(gate, upb, actb, Tt * Fc);
      cvt(wd + ((size_t)l * Ec + e) * Fc * Dd, wbuf, (size_t)Fc * Dd);
      gemmACC(actb, Fc, wbuf, Dd, x, Dd, Tt, Dd, Fc, 1.0f, wts + e, Ec);  // x += w_e*(act@Wd)
    }
  }

  // --- final norm + tied LM head ---
  rmsnorm_kernel<<<Tt, 256, 0, stream>>>(x, normo, hb);
  cvt(embedw, wbuf, (size_t)Vc * Dd);
  gemmBT(hb, Dd, wbuf, Dd, out, Vc, Tt, Vc, Dd, 1.0f);  // x @ embed^T
  write_aux_kernel<<<1, 1, 0, stream>>>(auxacc, out + (size_t)Tt * Vc);
}
